// AttentionMLP_72997264163220
// MI455X (gfx1250) — matrix operand order repaired and checked by reading, not compiled, not run on hardware
//
#include <hip/hip_runtime.h>
#include <hip/hip_bf16.h>

// ---------------- problem constants ----------------
#define D_IN   3800
#define DP     3840            // padded D (multiple of 64)
#define BATCH  16
#define SEQ    256
#define ROWS   (BATCH*SEQ)     // 4096
#define H_DIM  512
#define C_DIM  10
#define NBIG   (DP / 64)       // 60 big K-steps (64 halves each)

// ---------------- WMMA types ----------------
typedef __attribute__((ext_vector_type(8)))  float        v8f;
typedef __attribute__((ext_vector_type(16))) __bf16       v16bf;
typedef __attribute__((ext_vector_type(4)))  unsigned int uint4v;

union Frag { v16bf v; uint4v q[2]; };

__device__ __forceinline__ unsigned short f32_to_bf16(float f) {
    unsigned int u = __float_as_uint(f);
    u += 0x7FFFu + ((u >> 16) & 1u);     // round-to-nearest-even
    return (unsigned short)(u >> 16);
}
__device__ __forceinline__ float bf16_to_f32(unsigned short h) {
    return __uint_as_float(((unsigned int)h) << 16);
}

// ---------------- 1) convert + zero-pad f32 -> bf16 ----------------
__global__ void cvt_pad_kernel(const float* __restrict__ src,
                               unsigned short* __restrict__ dst,
                               int sRows, int sCols, int dCols) {
    int i = blockIdx.x * 256 + threadIdx.x;       // grid sized exactly dRows*dCols
    int r = i / dCols;
    int c = i - r * dCols;
    float v = (r < sRows && c < sCols) ? src[(size_t)r * sCols + c] : 0.0f;
    dst[i] = f32_to_bf16(v);
}

// ---------------- LDS-staged block GEMM core -----------------------
// Block tile: 128 rows x 64 features; 8 waves, wave w owns rows [w*16, w*16+16).
// B staging granule: 64 feature-rows x 64 K-halves (8KB), double-buffered in
// LDS (16KB). One barrier per granule covers two 32-K WMMA substeps (8 WMMAs).
// arow  : per-lane A row pointer (global, K-major, ld=DP)
// bcoop : per-thread cooperative B staging pointer (global); +32 for 2nd chunk
// lbs   : LDS, 2 * 64*64 halves
// sOff  : per-thread LDS store offset (halves); +32 for 2nd chunk
// rBase : per-lane fragment read base offset (halves)
__device__ __forceinline__ void gemm_block_tile(
    const unsigned short* __restrict__ arow,
    const unsigned short* __restrict__ bcoop,
    unsigned short* lbs, int sOff, int rBase,
    v8f acc[4]) {

    // preload big-step 0 into LDS buffer 0
    {
        uint4v g0 = *(const uint4v*)(bcoop);
        uint4v g1 = *(const uint4v*)(bcoop + 32);
        *(uint4v*)(lbs + sOff)      = g0;
        *(uint4v*)(lbs + sOff + 32) = g1;
    }
    __syncthreads();

#pragma unroll 1
    for (int s = 0; s < NBIG; ++s) {
        const int buf = s & 1;
        const unsigned short* lb = lbs + buf * (64 * 64) + rBase;
        const unsigned short* ag = arow + s * 64;

        // ---- substep 0: issue A + all B fragments, then prefetch, then WMMA
        Frag a0;
        a0.q[0] = *(const uint4v*)(ag);
        a0.q[1] = *(const uint4v*)(ag + 16);
        Frag b0[4];
#pragma unroll
        for (int n = 0; n < 4; ++n) {
            const unsigned short* lp = lb + n * (16 * 64);
            b0[n].q[0] = *(const uint4v*)lp;
            b0[n].q[1] = *(const uint4v*)(lp + 16);
        }

        // next big-step staging loads (drained only at the ds_store below)
        uint4v g0, g1;
        if (s + 1 < NBIG) {
            g0 = *(const uint4v*)(bcoop + (s + 1) * 64);
            g1 = *(const uint4v*)(bcoop + (s + 1) * 64 + 32);
        }

#pragma unroll
        for (int n = 0; n < 4; ++n)
            acc[n] = __builtin_amdgcn_wmma_f32_16x16x32_bf16(
                false, a0.v, false, b0[n].v, (short)0, acc[n], false, false);

        // ---- substep 1
        Frag a1;
        a1.q[0] = *(const uint4v*)(ag + 32);
        a1.q[1] = *(const uint4v*)(ag + 48);
        Frag b1[4];
#pragma unroll
        for (int n = 0; n < 4; ++n) {
            const unsigned short* lp = lb + n * (16 * 64) + 32;
            b1[n].q[0] = *(const uint4v*)lp;
            b1[n].q[1] = *(const uint4v*)(lp + 16);
        }
#pragma unroll
        for (int n = 0; n < 4; ++n)
            acc[n] = __builtin_amdgcn_wmma_f32_16x16x32_bf16(
                false, a1.v, false, b1[n].v, (short)0, acc[n], false, false);

        if (s + 1 < NBIG) {
            unsigned short* st = lbs + (buf ^ 1) * (64 * 64) + sOff;
            *(uint4v*)(st)      = g0;
            *(uint4v*)(st + 32) = g1;
        }
        __syncthreads();
    }
}

// ---------------- 2) QKV projection GEMM (bf16 WMMA, f32 acc) ------
// out[row, e] = sum_d x[row,d] * W[e,d] + bias[e]
__global__ void gemm_qkv_kernel(const unsigned short* __restrict__ xb,
                                const unsigned short* __restrict__ wk,
                                const unsigned short* __restrict__ wq,
                                const unsigned short* __restrict__ wv,
                                const float* __restrict__ bk,
                                const float* __restrict__ bq,
                                const float* __restrict__ bv,
                                unsigned short* __restrict__ ko,
                                unsigned short* __restrict__ qo,
                                unsigned short* __restrict__ vo) {
    __shared__ unsigned short lbs[2 * 64 * 64];   // 16KB double-buffered B tile

    const int which = blockIdx.y;                 // 0=k, 1=q, 2=v (uniform branch)
    const unsigned short* w    = (which == 0) ? wk : (which == 1) ? wq : wv;
    const float*          bias = (which == 0) ? bk : (which == 1) ? bq : bv;
    unsigned short*       out  = (which == 0) ? ko : (which == 1) ? qo : vo;

    const int tid  = threadIdx.x;
    const int wave = tid >> 5;
    const int lane = tid & 31;
    const int half = lane >> 4;
    const int m    = lane & 15;

    const int rowBlock = blockIdx.x / 60;         // 32 row blocks of 128 rows
    const int ft       = blockIdx.x - rowBlock * 60;
    const int row0     = rowBlock * 128 + wave * 16;
    const int e0       = ft * 64;

    const unsigned short* arow  = xb + (size_t)(row0 + m) * DP + (half ? 8 : 0);
    const unsigned short* bcoop = w  + (size_t)(e0 + (tid >> 2)) * DP + (tid & 3) * 8;
    const int sOff  = (tid >> 2) * 64 + (tid & 3) * 8;
    const int rBase = m * 64 + half * 8;

    v8f acc[4] = {v8f{}, v8f{}, v8f{}, v8f{}};
    gemm_block_tile(arow, bcoop, lbs, sOff, rBase, acc);

    // C layout: VGPR r -> row (r + half*8), N = m
#pragma unroll
    for (int n = 0; n < 4; ++n) {
        const int e = e0 + n * 16 + m;
        const float bias_e = (e < D_IN) ? bias[e] : 0.0f;
        unsigned short* orow = out + (size_t)(row0 + half * 8) * DP + e;
#pragma unroll
        for (int r = 0; r < 8; ++r) {
            orow[(size_t)r * DP] = f32_to_bf16(acc[n][r] + bias_e);
        }
    }
}

// ---------------- 3) scores GEMM: q @ k^T / sqrt(D) ----------------
__global__ void gemm_scores_kernel(const unsigned short* __restrict__ qb,
                                   const unsigned short* __restrict__ kb,
                                   float* __restrict__ attn) {
    __shared__ unsigned short lbs[2 * 64 * 64];
    const float scale = 0.016222142113076254f;    // 1/sqrt(3800)

    const int tid  = threadIdx.x;
    const int wave = tid >> 5;
    const int lane = tid & 31;
    const int half = lane >> 4;
    const int m    = lane & 15;

    const int b  = blockIdx.x >> 3;               // batch (128 blocks total)
    const int r  = blockIdx.x & 7;
    const int i0 = (r >> 2) * 128 + wave * 16;    // 2 row blocks of 128 queries
    const int j0 = (r & 3) * 64;                  // 4 key-64 tiles

    const unsigned short* qbase = qb + (size_t)b * SEQ * DP;
    const unsigned short* kbase = kb + (size_t)b * SEQ * DP;
    const unsigned short* arow  = qbase + (size_t)(i0 + m) * DP + (half ? 8 : 0);
    const unsigned short* bcoop = kbase + (size_t)(j0 + (tid >> 2)) * DP + (tid & 3) * 8;
    const int sOff  = (tid >> 2) * 64 + (tid & 3) * 8;
    const int rBase = m * 64 + half * 8;

    v8f acc[4] = {v8f{}, v8f{}, v8f{}, v8f{}};
    gemm_block_tile(arow, bcoop, lbs, sOff, rBase, acc);

    float* abase = attn + (size_t)b * SEQ * SEQ;
#pragma unroll
    for (int n = 0; n < 4; ++n) {
        const int j = j0 + n * 16 + m;
#pragma unroll
        for (int r8 = 0; r8 < 8; ++r8) {
            const int i = i0 + r8 + half * 8;
            abase[(size_t)i * SEQ + j] = acc[n][r8] * scale;
        }
    }
}

// ---------------- 4) row softmax over 256 (wave per row) -----------
__global__ void softmax_kernel(float* __restrict__ attn) {
    const int wave = threadIdx.x >> 5;
    const int lane = threadIdx.x & 31;
    const int row  = blockIdx.x * 8 + wave;       // 0..4095
    float* p = attn + (size_t)row * SEQ;

    float v[8];
    float mx = -3.4e38f;
#pragma unroll
    for (int t = 0; t < 8; ++t) { v[t] = p[lane + 32 * t]; mx = fmaxf(mx, v[t]); }
    for (int off = 16; off > 0; off >>= 1) mx = fmaxf(mx, __shfl_xor(mx, off, 32));
    float sum = 0.0f;
#pragma unroll
    for (int t = 0; t < 8; ++t) { v[t] = __expf(v[t] - mx); sum += v[t]; }
    for (int off = 16; off > 0; off >>= 1) sum += __shfl_xor(sum, off, 32);
    const float inv = 1.0f / sum;
#pragma unroll
    for (int t = 0; t < 8; ++t) p[lane + 32 * t] = v[t] * inv;
}

// ---------------- 5) column-mean of attn: w[b,j] -------------------
__global__ void colmean_kernel(const float* __restrict__ attn,
                               float* __restrict__ wmean) {
    const int i = blockIdx.x * 256 + threadIdx.x; // 0..4095
    const int b = i >> 8;
    const int j = i & 255;
    const float* base = attn + (size_t)b * SEQ * SEQ + j;
    float s = 0.0f;
    for (int q = 0; q < SEQ; ++q) s += base[(size_t)q * SEQ];
    wmean[i] = s * (1.0f / (float)SEQ);
}

// ---------------- 6) h[b,d] = sum_j w[b,j] * v[b,j,d] --------------
__global__ void hpool_kernel(const unsigned short* __restrict__ vb,
                             const float* __restrict__ wmean,
                             float* __restrict__ h) {
    __shared__ float wr[SEQ];
    const int b   = blockIdx.y;
    const int tid = threadIdx.x;
    wr[tid] = wmean[b * SEQ + tid];
    __syncthreads();
    const int d = blockIdx.x * 256 + tid;
    if (d < D_IN) {
        const unsigned short* vbase = vb + (size_t)b * SEQ * DP + d;
        float s = 0.0f;
        for (int j = 0; j < SEQ; ++j) s += wr[j] * bf16_to_f32(vbase[(size_t)j * DP]);
        h[(size_t)b * D_IN + d] = s;
    }
}

// ---------------- 7) MLP layer: wave-per-output dot ----------------
__global__ void mlp_kernel(const float* __restrict__ in,
                           const float* __restrict__ W,
                           const float* __restrict__ bias,
                           float* __restrict__ out,
                           int K, int O, int act) {
    const int wave = threadIdx.x >> 5;
    const int lane = threadIdx.x & 31;
    const int w = blockIdx.x * 8 + wave;
    const int b = w / O;
    const int o = w - b * O;
    const float* ip = in + (size_t)b * K;
    const float* wp = W  + (size_t)o * K;
    float s = 0.0f;
    for (int k = lane; k < K; k += 32) s += ip[k] * wp[k];
    for (int off = 16; off > 0; off >>= 1) s += __shfl_xor(s, off, 32);
    if (lane == 0) {
        float r = s + bias[o];
        if (act) r = 1.0f / (1.0f + __expf(-r));
        out[(size_t)b * O + o] = r;
    }
}

// ---------------- launch ----------------
extern "C" void kernel_launch(void* const* d_in, const int* in_sizes, int n_in,
                              void* d_out, int out_size, void* d_ws, size_t ws_size,
                              hipStream_t stream) {
    const float* x  = (const float*)d_in[0];
    const float* Wk = (const float*)d_in[1];
    const float* bk = (const float*)d_in[2];
    const float* Wq = (const float*)d_in[3];
    const float* bq = (const float*)d_in[4];
    const float* Wv = (const float*)d_in[5];
    const float* bv = (const float*)d_in[6];
    const float* W1 = (const float*)d_in[7];
    const float* b1 = (const float*)d_in[8];
    const float* W2 = (const float*)d_in[9];
    const float* b2 = (const float*)d_in[10];
    const float* W3 = (const float*)d_in[11];
    const float* b3 = (const float*)d_in[12];
    float* out = (float*)d_out;

    // workspace layout (all region sizes are multiples of 256 bytes)
    char* ws = (char*)d_ws;
    size_t off = 0;
    const size_t SZ_XB = (size_t)ROWS * DP * 2;
    const size_t SZ_WB = (size_t)DP   * DP * 2;
    const size_t SZ_QB = (size_t)BATCH * SEQ * DP * 2;
    unsigned short* xb  = (unsigned short*)(ws + off); off += SZ_XB;
    unsigned short* wkb = (unsigned short*)(ws + off); off += SZ_WB;
    unsigned short* wqb = (unsigned short*)(ws + off); off += SZ_WB;
    unsigned short* wvb = (unsigned short*)(ws + off); off += SZ_WB;
    unsigned short* qb  = (unsigned short*)(ws + off); off += SZ_QB;
    unsigned short* kb  = (unsigned short*)(ws + off); off += SZ_QB;
    unsigned short* vb  = (unsigned short*)(ws + off); off += SZ_QB;
    float* attn  = (float*)(ws + off); off += (size_t)BATCH * SEQ * SEQ * 4;
    float* wmean = (float*)(ws + off); off += (size_t)BATCH * SEQ * 4;
    float* h     = (float*)(ws + off); off += (size_t)BATCH * D_IN * 4;
    float* h1    = (float*)(ws + off); off += (size_t)BATCH * H_DIM * 4;
    float* h2    = (float*)(ws + off); off += (size_t)BATCH * H_DIM * 4;

    // 1) bf16 convert + pad
    cvt_pad_kernel<<<(ROWS * DP) / 256, 256, 0, stream>>>(x,  xb,  ROWS,  D_IN, DP);
    cvt_pad_kernel<<<(DP   * DP) / 256, 256, 0, stream>>>(Wk, wkb, D_IN, D_IN, DP);
    cvt_pad_kernel<<<(DP   * DP) / 256, 256, 0, stream>>>(Wq, wqb, D_IN, D_IN, DP);
    cvt_pad_kernel<<<(DP   * DP) / 256, 256, 0, stream>>>(Wv, wvb, D_IN, D_IN, DP);

    // 2) QKV projections: 32 row-blocks x 60 feature tiles, y over {k,q,v}
    gemm_qkv_kernel<<<dim3(1920, 3, 1), 256, 0, stream>>>(
        xb, wkb, wqb, wvb, bk, bq, bv, kb, qb, vb);

    // 3) attention scores: 16 batches x (2 row-blocks x 4 key tiles)
    gemm_scores_kernel<<<128, 256, 0, stream>>>(qb, kb, attn);

    // 4) softmax over rows (4096 rows, 8 waves/block)
    softmax_kernel<<<512, 256, 0, stream>>>(attn);

    // 5) column mean of attn
    colmean_kernel<<<16, 256, 0, stream>>>(attn, wmean);

    // 6) pooled h = colmean(attn) @ v
    hpool_kernel<<<dim3((D_IN + 255) / 256, BATCH, 1), 256, 0, stream>>>(vb, wmean, h);

    // 7) MLP head
    mlp_kernel<<<(BATCH * H_DIM) / 8, 256, 0, stream>>>(h,  W1, b1, h1, D_IN,  H_DIM, 1);
    mlp_kernel<<<(BATCH * H_DIM) / 8, 256, 0, stream>>>(h1, W2, b2, h2, H_DIM, H_DIM, 1);
    mlp_kernel<<<(BATCH * C_DIM) / 8, 256, 0, stream>>>(h2, W3, b3, out, H_DIM, C_DIM, 0);
}
